// MultiheadAttention_39977555591771
// MI455X (gfx1250) — matrix operand order repaired and checked by reading, not compiled, not run on hardware
//
#include <hip/hip_runtime.h>

// ---------------------------------------------------------------------------
// MHA for MI455X (gfx1250): f16 WMMA (f32 accum) GEMMs + flash-attention with
// TDM (tensor_load_to_lds) double-buffered K/V tiles and async-LDS staging.
//   d_in: q,k,v,mask,wq,bq,wk,bk,wv,bv,wo,bo   d_out: [4096,1024] f32
// ---------------------------------------------------------------------------

typedef _Float16 half_t;
typedef __attribute__((ext_vector_type(16))) _Float16 v16h;
typedef __attribute__((ext_vector_type(8)))  float    v8f;
typedef __attribute__((ext_vector_type(4)))  unsigned u32x4;
typedef __attribute__((ext_vector_type(4)))  int      i32x4;
typedef __attribute__((ext_vector_type(8)))  int      i32x8;

constexpr int SEQ = 4096;
constexpr int DM  = 1024;
constexpr int NHEAD = 16;
constexpr int DKH = 64;
constexpr size_t HS = (size_t)NHEAD * SEQ * DKH; // halfs per head-major tensor

#define WMMA_F32_F16(A, B, C) \
  __builtin_amdgcn_wmma_f32_16x16x32_f16(false, (A), false, (B), (short)0, (C), false, false)

static __device__ inline v16h frag2(const half_t* p0, const half_t* p1) {
  v16h f;
  ((uint4*)&f)[0] = *(const uint4*)p0;
  ((uint4*)&f)[1] = *(const uint4*)p1;
  return f;
}

// Async copy 16 bytes/lane from global to LDS (ASYNCcnt-tracked, GV mode).
static __device__ inline void async_lds_b128(void* lds_ptr, const void* gptr) {
  unsigned lds = (unsigned)(size_t)lds_ptr;
  unsigned long long ga = (unsigned long long)(size_t)gptr;
  asm volatile("global_load_async_to_lds_b128 %0, %1, off"
               :: "v"(lds), "v"(ga)
               : "memory");
}
static __device__ inline void wait_async0() {
  asm volatile("s_wait_asynccnt 0x0" ::: "memory");
}

// ---------------------------------------------------------------------------
// TDM: 2-D tile load global->LDS.  data_size = 8B units.  Hardware padding:
// 16B of LDS pad after every 128B row => LDS row stride 144B (72 halfs),
// matching the bank-conflict-free fragment layout used by the WMMA loops.
//   rowlen8  : tile/tensor dim0 in 8-byte units (16 => 128B rows)
//   nrows    : tile/tensor dim1 (rows)
//   stride8  : tensor dim0 stride in 8-byte units
// ---------------------------------------------------------------------------
static __device__ inline void tdm_load_2d(void* lds_ptr, const void* gptr,
                                          unsigned rowlen8, unsigned nrows,
                                          unsigned stride8) {
  unsigned long long ga = (unsigned long long)(size_t)gptr;
  u32x4 g0;
  g0[0] = 1u;                                   // count=1, user descriptor
  g0[1] = (unsigned)(size_t)lds_ptr;            // lds_addr (bytes)
  g0[2] = (unsigned)ga;                         // global_addr[31:0]
  g0[3] = (unsigned)((ga >> 32) & 0x1FFFFFFu)   // global_addr[56:32]
          | (2u << 30);                         // type = 2 ("image")
  i32x8 g1 = {};
  g1[0] = (3 << 16)        // data_size = 8B
        | (1 << 20)        // pad_enable
        | (4 << 22)        // pad_interval: 32 DWORDs (128B)
        | (3 << 25);       // pad_amount:   4 DWORDs (16B)
  g1[1] = (int)(rowlen8 << 16);                 // tensor_dim0[15:0]
  g1[2] = (int)((rowlen8 >> 16) | (nrows << 16)); // dim0 hi | tensor_dim1 lo
  g1[3] = (int)((nrows >> 16) | (rowlen8 << 16)); // dim1 hi | tile_dim0
  g1[4] = (int)(nrows & 0xFFFFu);               // tile_dim1 | tile_dim2=0
  g1[5] = (int)stride8;                         // tensor_dim0_stride[31:0]
  g1[6] = 0;                                    // stride hi | dim1_stride lo
  g1[7] = 0;
  i32x4 z4 = {};
#if defined(__clang_major__) && __clang_major__ >= 23
  i32x8 z8 = {};
  __builtin_amdgcn_tensor_load_to_lds(g0, g1, z4, z4, z8, 0);
#else
  __builtin_amdgcn_tensor_load_to_lds(g0, g1, z4, z4, 0);
#endif
}

// ---------------------------------------------------------------------------
// Kernel 1: fused QKV projection.  y = x @ W^T + b, output f16 head-major.
// blockIdx.z: 0 -> Qh[h][s][dk], 1 -> Kh[h][s][dk], 2 -> Vt[h][dk][s]
// Block: 256 threads = 8 waves; tile 128(M) x 64(N); K-step 32.
// ---------------------------------------------------------------------------
__global__ __launch_bounds__(256) void qkv_proj_kernel(
    const float* __restrict__ xq, const float* __restrict__ xk, const float* __restrict__ xv,
    const float* __restrict__ wqp, const float* __restrict__ wkp, const float* __restrict__ wvp,
    const float* __restrict__ bqp, const float* __restrict__ bkp, const float* __restrict__ bvp,
    half_t* __restrict__ Qh, half_t* __restrict__ Kh, half_t* __restrict__ Vt)
{
  const int which = blockIdx.z;
  const float* X = (which == 0) ? xq : (which == 1) ? xk : xv;
  const float* W = (which == 0) ? wqp : (which == 1) ? wkp : wvp;
  const float* Bv = (which == 0) ? bqp : (which == 1) ? bkp : bvp;
  half_t* Out = (which == 0) ? Qh : (which == 1) ? Kh : Vt;

  const int m0 = blockIdx.y * 128;
  const int n0 = blockIdx.x * 64;       // one head's dk range (64 | head)
  const int head = n0 >> 6;
  const int t = threadIdx.x;
  const int lane = t & 31, wave = t >> 5;
  const int hi = lane >> 4, ln = lane & 15;

  __shared__ alignas(16) half_t As[128][40];   // [m][k], padded
  __shared__ alignas(16) half_t Bs[64][40];    // [n][k] == rows of W, padded

  v8f acc[4] = {};

  for (int k0 = 0; k0 < DM; k0 += 32) {
    // stage A tile (fp32 -> fp16): 128x32, 16 floats per thread
    {
      const int r = t >> 1, kc = (t & 1) * 16;
      const float4* src = (const float4*)(X + (size_t)(m0 + r) * DM + k0 + kc);
#pragma unroll
      for (int j = 0; j < 4; ++j) {
        float4 f = src[j];
        As[r][kc + 4 * j + 0] = (half_t)f.x;
        As[r][kc + 4 * j + 1] = (half_t)f.y;
        As[r][kc + 4 * j + 2] = (half_t)f.z;
        As[r][kc + 4 * j + 3] = (half_t)f.w;
      }
    }
    // stage B tile: W rows n0..n0+63, 8 floats per thread
    {
      const int r = t >> 2, kc = (t & 3) * 8;
      const float4* src = (const float4*)(W + (size_t)(n0 + r) * DM + k0 + kc);
#pragma unroll
      for (int j = 0; j < 2; ++j) {
        float4 f = src[j];
        Bs[r][kc + 4 * j + 0] = (half_t)f.x;
        Bs[r][kc + 4 * j + 1] = (half_t)f.y;
        Bs[r][kc + 4 * j + 2] = (half_t)f.z;
        Bs[r][kc + 4 * j + 3] = (half_t)f.w;
      }
    }
    __syncthreads();

    const int kb = 8 * hi;
    v16h af = frag2(&As[wave * 16 + ln][kb], &As[wave * 16 + ln][kb + 16]);
#pragma unroll
    for (int nt = 0; nt < 4; ++nt) {
      const int bn = nt * 16 + ln;      // N column == W row
      const int bkb = hi * 8;           // K 8h..8h+7 and 16+8h..16+8h+7 per lane half
      v16h bf = frag2(&Bs[bn][bkb], &Bs[bn][bkb + 16]);
      acc[nt] = WMMA_F32_F16(af, bf, acc[nt]);
    }
    __syncthreads();
  }

  // epilogue: bias add, fp16 convert, head-major store
#pragma unroll
  for (int nt = 0; nt < 4; ++nt) {
    const int col = nt * 16 + ln;       // dk within head
    const float b = Bv[n0 + col];
#pragma unroll
    for (int i = 0; i < 8; ++i) {
      const int gm = m0 + wave * 16 + i + 8 * hi;   // sequence position
      const half_t h = (half_t)(acc[nt][i] + b);
      if (which < 2)
        Out[((size_t)head * SEQ + gm) * DKH + col] = h;          // [h][s][dk]
      else
        Out[((size_t)head * DKH + col) * SEQ + gm] = h;          // [h][dk][s]
    }
  }
}

// ---------------------------------------------------------------------------
// Kernel 2: flash attention per head, TDM double-buffered K/V tiles.
// Block: 256 threads = 8 waves; 128 queries per block; kv chunk = 64.
// Wave 0 drives the TDM engine; TENSORcnt wait + barrier publishes tiles.
// ---------------------------------------------------------------------------
__global__ __launch_bounds__(256) void attn_kernel(
    const half_t* __restrict__ Qh, const half_t* __restrict__ Kh,
    const half_t* __restrict__ Vt, const unsigned char* __restrict__ mask,
    half_t* __restrict__ Concat)
{
  const int h = blockIdx.y;
  const int q0 = blockIdx.x * 128;
  const int t = threadIdx.x;
  const int lane = t & 31, wave = t >> 5;
  const int hi = lane >> 4, ln = lane & 15;
  const int qw = q0 + wave * 16;        // this wave's 16 query rows

  __shared__ alignas(16) half_t Ks[2][64][72];  // [buf][kpos][d] (B-frags QK^T)
  __shared__ alignas(16) half_t Vs[2][64][72];  // [buf][d][kpos] (B-frags P.V)
  __shared__ alignas(16) half_t Ps[8][16][72];  // per-wave P staging [m][kpos]

  // Q fragments (kept resident): rows qw..qw+15, d = 0..63 -> 2 A-frags
  v16h qa[2];
  {
    const half_t* qrow = Qh + ((size_t)h * SEQ + qw + ln) * DKH;
#pragma unroll
    for (int kk = 0; kk < 2; ++kk) {
      const int base = kk * 32 + 8 * hi;
      qa[kk] = frag2(qrow + base, qrow + base + 16);
    }
  }

  float mrun[8], lrun[8];
#pragma unroll
  for (int i = 0; i < 8; ++i) { mrun[i] = -3.0e38f; lrun[i] = 0.0f; }
  v8f accO[4] = {};
  const float scale = 0.03125f;         // 1/sqrt(1024)

  // prologue: TDM-stage tile 0 into buffer 0 (wave 0 drives the DMA)
  if (wave == 0) {
    tdm_load_2d(&Ks[0][0][0], Kh + ((size_t)h * SEQ) * DKH, 16, 64, 16);
    tdm_load_2d(&Vs[0][0][0], Vt + ((size_t)h * DKH) * SEQ, 16, 64, SEQ / 4);
  }

  int cur = 0;
  for (int kv0 = 0; kv0 < SEQ; kv0 += 64) {
    if (wave == 0) __builtin_amdgcn_s_wait_tensorcnt(0);  // tile(kv0) landed
    __syncthreads();   // publish tile; all waves past tile(kv0-64) compute

    if (wave == 0 && kv0 + 64 < SEQ) {  // DMA next tile into other buffer
      const int nb = cur ^ 1;
      tdm_load_2d(&Ks[nb][0][0], Kh + ((size_t)h * SEQ + kv0 + 64) * DKH, 16, 64, 16);
      tdm_load_2d(&Vs[nb][0][0], Vt + ((size_t)h * DKH) * SEQ + kv0 + 64, 16, 64, SEQ / 4);
    }

    // S = scale * Q K^T, masked
    float sv[4][8];
#pragma unroll
    for (int nt = 0; nt < 4; ++nt) {
      v8f sacc = {};
#pragma unroll
      for (int kk = 0; kk < 2; ++kk) {
        const int bkb = kk * 32 + hi * 8;
        v16h bf = frag2(&Ks[cur][nt * 16 + ln][bkb], &Ks[cur][nt * 16 + ln][bkb + 16]);
        sacc = WMMA_F32_F16(qa[kk], bf, sacc);
      }
      const int kn = kv0 + nt * 16 + ln;
#pragma unroll
      for (int i = 0; i < 8; ++i) {
        const int qm = qw + i + 8 * hi;
        float s = sacc[i] * scale;
        if (mask[(size_t)qm * SEQ + kn]) s = -1.0e9f;
        sv[nt][i] = s;
      }
    }

    // running softmax stats (row lives in one 16-lane half)
    float mnew[8], alpha[8], rsum[8];
#pragma unroll
    for (int i = 0; i < 8; ++i) {
      float mx = fmaxf(fmaxf(sv[0][i], sv[1][i]), fmaxf(sv[2][i], sv[3][i]));
#pragma unroll
      for (int off = 1; off < 16; off <<= 1)
        mx = fmaxf(mx, __shfl_xor(mx, off, 16));
      mnew[i] = fmaxf(mrun[i], mx);
      alpha[i] = __expf(mrun[i] - mnew[i]);
      rsum[i] = 0.0f;
    }
    // P = exp(S - m), stage to LDS in A-frag source layout [m][kpos]
#pragma unroll
    for (int nt = 0; nt < 4; ++nt)
#pragma unroll
      for (int i = 0; i < 8; ++i) {
        float p = __expf(sv[nt][i] - mnew[i]);
        rsum[i] += p;
        Ps[wave][i + 8 * hi][nt * 16 + ln] = (half_t)p;
      }
#pragma unroll
    for (int i = 0; i < 8; ++i) {
#pragma unroll
      for (int off = 1; off < 16; off <<= 1)
        rsum[i] += __shfl_xor(rsum[i], off, 16);
      lrun[i] = lrun[i] * alpha[i] + rsum[i];
      mrun[i] = mnew[i];
    }
#pragma unroll
    for (int nt = 0; nt < 4; ++nt)
#pragma unroll
      for (int i = 0; i < 8; ++i) accO[nt][i] *= alpha[i];

    __syncthreads();  // Ps visible across the wave's lanes

    // O += P V
    v16h pa[2];
#pragma unroll
    for (int kk = 0; kk < 2; ++kk) {
      const int base = kk * 32 + 8 * hi;
      pa[kk] = frag2(&Ps[wave][ln][base], &Ps[wave][ln][base + 16]);
    }
#pragma unroll
    for (int nt = 0; nt < 4; ++nt)
#pragma unroll
      for (int kk = 0; kk < 2; ++kk) {
        const int bkb = kk * 32 + hi * 8;
        v16h bf = frag2(&Vs[cur][nt * 16 + ln][bkb], &Vs[cur][nt * 16 + ln][bkb + 16]);
        accO[nt] = WMMA_F32_F16(pa[kk], bf, accO[nt]);
      }
    cur ^= 1;
  }

  // finalize: O / l, write concat f16 [s][h*64+d]
#pragma unroll
  for (int nt = 0; nt < 4; ++nt) {
    const int d = nt * 16 + ln;
#pragma unroll
    for (int i = 0; i < 8; ++i) {
      const int qm = qw + i + 8 * hi;
      Concat[(size_t)qm * DM + h * DKH + d] = (half_t)(accO[nt][i] / lrun[i]);
    }
  }
}

// ---------------------------------------------------------------------------
// Kernel 3: output projection.  out = concat(f16) @ wo^T + bo  (fp32 out)
// A tile streamed async to LDS (already f16); W tile converted via registers.
// ---------------------------------------------------------------------------
__global__ __launch_bounds__(256) void out_proj_kernel(
    const half_t* __restrict__ Concat, const float* __restrict__ wo,
    const float* __restrict__ bo, float* __restrict__ out)
{
  const int m0 = blockIdx.y * 128;
  const int n0 = blockIdx.x * 64;
  const int t = threadIdx.x;
  const int lane = t & 31, wave = t >> 5;
  const int hi = lane >> 4, ln = lane & 15;

  __shared__ alignas(16) half_t As[128][40];
  __shared__ alignas(16) half_t Bs[64][40];

  v8f acc[4] = {};

  for (int k0 = 0; k0 < DM; k0 += 32) {
    {
      const int r = t >> 1, kc = (t & 1) * 16;
      const half_t* src = Concat + (size_t)(m0 + r) * DM + k0 + kc;
      async_lds_b128(&As[r][kc],     src);
      async_lds_b128(&As[r][kc + 8], src + 8);
    }
    {
      const int r = t >> 2, kc = (t & 3) * 8;
      const float4* src = (const float4*)(wo + (size_t)(n0 + r) * DM + k0 + kc);
#pragma unroll
      for (int j = 0; j < 2; ++j) {
        float4 f = src[j];
        Bs[r][kc + 4 * j + 0] = (half_t)f.x;
        Bs[r][kc + 4 * j + 1] = (half_t)f.y;
        Bs[r][kc + 4 * j + 2] = (half_t)f.z;
        Bs[r][kc + 4 * j + 3] = (half_t)f.w;
      }
    }
    wait_async0();
    __syncthreads();

    const int kb = 8 * hi;
    v16h af = frag2(&As[wave * 16 + ln][kb], &As[wave * 16 + ln][kb + 16]);
#pragma unroll
    for (int nt = 0; nt < 4; ++nt) {
      const int bkb = hi * 8;
      v16h bf = frag2(&Bs[nt * 16 + ln][bkb], &Bs[nt * 16 + ln][bkb + 16]);
      acc[nt] = WMMA_F32_F16(af, bf, acc[nt]);
    }
    __syncthreads();
  }

#pragma unroll
  for (int nt = 0; nt < 4; ++nt) {
    const int n = n0 + nt * 16 + ln;
    const float b = bo[n];
#pragma unroll
    for (int i = 0; i < 8; ++i) {
      const int gm = m0 + wave * 16 + i + 8 * hi;
      out[(size_t)gm * DM + n] = acc[nt][i] + b;
    }
  }
}

// ---------------------------------------------------------------------------
extern "C" void kernel_launch(void* const* d_in, const int* in_sizes, int n_in,
                              void* d_out, int out_size, void* d_ws, size_t ws_size,
                              hipStream_t stream) {
  (void)in_sizes; (void)n_in; (void)out_size; (void)ws_size;

  const float* q  = (const float*)d_in[0];
  const float* k  = (const float*)d_in[1];
  const float* v  = (const float*)d_in[2];
  const unsigned char* mask = (const unsigned char*)d_in[3];
  const float* wq = (const float*)d_in[4];
  const float* bq = (const float*)d_in[5];
  const float* wk = (const float*)d_in[6];
  const float* bk = (const float*)d_in[7];
  const float* wv = (const float*)d_in[8];
  const float* bv = (const float*)d_in[9];
  const float* wo = (const float*)d_in[10];
  const float* bo = (const float*)d_in[11];
  float* out = (float*)d_out;

  half_t* ws = (half_t*)d_ws;
  half_t* Qh = ws;               // [h][s][dk]  8 MB
  half_t* Kh = ws + HS;          // [h][s][dk]  8 MB
  half_t* Vt = ws + 2 * HS;      // [h][dk][s]  8 MB
  half_t* Cc = ws + 3 * HS;      // [s][dm]     8 MB

  qkv_proj_kernel<<<dim3(DM / 64, SEQ / 128, 3), 256, 0, stream>>>(
      q, k, v, wq, wk, wv, bq, bk, bv, Qh, Kh, Vt);
  attn_kernel<<<dim3(SEQ / 128, NHEAD), 256, 0, stream>>>(Qh, Kh, Vt, mask, Cc);
  out_proj_kernel<<<dim3(DM / 64, SEQ / 128), 256, 0, stream>>>(Cc, wo, bo, out);
}
